// DecoderAttention_77515569759043
// MI455X (gfx1250) — matrix operand order repaired and checked by reading, not compiled, not run on hardware
//
#include <hip/hip_runtime.h>
#include <hip/hip_bf16.h>
#include <stdint.h>

// ---------------------------------------------------------------------------
// DecoderAttention for MI455X (gfx1250, wave32, WMMA 16x16x32 f16/f32-acc,
// async global->LDS staging in the fused attention kernel).
// B=8 T=1024 N=512 L=4096 D=512 F=64 BR=8
// ---------------------------------------------------------------------------

typedef __attribute__((ext_vector_type(16))) _Float16 v16h;
typedef __attribute__((ext_vector_type(8)))  _Float16 v8h;
typedef __attribute__((ext_vector_type(8)))  float    v8f;

#define DA_B  8
#define DA_T  1024
#define DA_N  512
#define DA_L  4096
#define DA_D  512
#define DA_F  64
#define DA_BR 8

__device__ __forceinline__ v16h da_cat8(v8h lo, v8h hi) {
  return __builtin_shufflevector(lo, hi, 0,1,2,3,4,5,6,7,8,9,10,11,12,13,14,15);
}
__device__ __forceinline__ v8f da_wmma(v16h a, v16h b, v8f c) {
  // D = A(16xK=32) * B(K=32x16) + C, f32 accumulate
  return __builtin_amdgcn_wmma_f32_16x16x32_f16(false, a, false, b, (short)0, c,
                                                false, false);
}

// ---- CDNA5 async global->LDS (ASYNCcnt-tracked DMA, 16 B per lane) --------
__device__ __forceinline__ void da_async_b128(uint32_t lds_off, const void* gaddr) {
  asm volatile("global_load_async_to_lds_b128 %0, %1, off"
               :: "v"(lds_off), "v"(gaddr) : "memory");
}
__device__ __forceinline__ void da_wait_async0() {
  asm volatile("s_wait_asynccnt 0x0" ::: "memory");
}
__device__ __forceinline__ uint32_t da_lds_off(const void* p) {
  // Generic LDS addresses map to LDS by truncation to the low 32 bits.
  return (uint32_t)(uintptr_t)p;
}

// ---------------------------------------------------------------------------
// Weight convert+transpose: W[D,F] f32 -> WT[F,D] f16
// ---------------------------------------------------------------------------
__global__ void da_wcvt_kernel(const float* __restrict__ W, _Float16* __restrict__ WT) {
  int i = blockIdx.x * blockDim.x + threadIdx.x;        // 0 .. D*F-1
  int k = i >> 6;                                       // D index
  int f = i & 63;                                       // F index
  WT[(size_t)f * DA_D + k] = (_Float16)W[i];
}

// ---------------------------------------------------------------------------
// Projection GEMM: Y[rows,F] = X[rows,D] @ WT^T + bias.
// One wave per 16-row tile; K swept in 32-chunks -> 16*4 = 64 WMMAs per tile.
// Optional outputs: f32 row-major, f16 row-major, f16 transposed [B,F,DA_L].
// ---------------------------------------------------------------------------
__global__ void __launch_bounds__(256)
da_proj_kernel(const float* __restrict__ X,
               const _Float16* __restrict__ WT,
               const float* __restrict__ bias,
               float* __restrict__ out32,
               _Float16* __restrict__ out16,
               _Float16* __restrict__ out16T,   // [B, F, DA_L] (rows/batch == DA_L)
               int rows) {
  int wave = (blockIdx.x * blockDim.x + threadIdx.x) >> 5;
  int lane = threadIdx.x & 31;
  int li = lane & 15;
  int hf = lane >> 4;
  int rowbase = wave * 16;
  if (rowbase >= rows) return;

  const float* xr = X + (size_t)(rowbase + li) * DA_D + hf * 16;
  v8f c[4];
  c[0] = {}; c[1] = {}; c[2] = {}; c[3] = {};

  for (int kc = 0; kc < DA_D; kc += 32) {
    // A operand: 16 rows of X, K-chunk [kc, kc+32)  (f32 -> f16 on the fly)
    v8f xlo = *(const v8f*)(xr + kc);
    v8f xhi = *(const v8f*)(xr + kc + 8);
    v16h a = da_cat8(__builtin_convertvector(xlo, v8h),
                     __builtin_convertvector(xhi, v8h));
#pragma unroll
    for (int g = 0; g < 4; ++g) {
      // B operand: WT row = output feature (col), contiguous K
      v16h bf = *(const v16h*)(WT + (size_t)(g * 16 + li) * DA_D + kc + hf * 16);
      c[g] = da_wmma(a, bf, c[g]);
    }
  }

#pragma unroll
  for (int g = 0; g < 4; ++g) {
    float bv = bias[g * 16 + li];
#pragma unroll
    for (int j = 0; j < 8; ++j) {
      int R = rowbase + j + 8 * hf;    // absolute row
      int f = g * 16 + li;             // feature
      float val = c[g][j] + bv;
      size_t o = (size_t)R * DA_F + f;
      if (out32) out32[o] = val;
      if (out16) out16[o] = (_Float16)val;
      if (out16T) {
        int bb = R >> 12;              // R / DA_L
        int lr = R & (DA_L - 1);       // R % DA_L
        out16T[((size_t)bb * DA_F + f) * DA_L + lr] = (_Float16)val;
      }
    }
  }
}

// ---------------------------------------------------------------------------
// Aggregation logits: (leaf @ Wagg + bagg) -> [B*L]. One wave per row.
// ---------------------------------------------------------------------------
__global__ void da_agg_logits_kernel(const float* __restrict__ leaf,
                                     const float* __restrict__ Wagg,
                                     const float* __restrict__ bagg,
                                     float* __restrict__ logits) {
  int wave = (blockIdx.x * blockDim.x + threadIdx.x) >> 5;
  int lane = threadIdx.x & 31;
  if (wave >= DA_B * DA_L) return;
  const float* xr = leaf + (size_t)wave * DA_D;
  float sum = 0.f;
#pragma unroll
  for (int i = 0; i < DA_D / 32; ++i) sum += xr[lane + 32 * i] * Wagg[lane + 32 * i];
#pragma unroll
  for (int off = 16; off; off >>= 1) sum += __shfl_xor(sum, off);
  if (lane == 0) logits[wave] = sum + bagg[0];
}

// ---------------------------------------------------------------------------
// interp = (root + node_v[parent] + leaf_v)/3 ; upward = suffix-mean over L.
// One block per batch, thread = feature (coalesced 64-wide rows).
// ---------------------------------------------------------------------------
__global__ void da_upward_kernel(const float* __restrict__ root,
                                 const float* __restrict__ nodev,
                                 const float* __restrict__ leafv,
                                 float* __restrict__ upward) {
  int b = blockIdx.x;
  int f = threadIdx.x;                       // 0..63
  float r = root[b * DA_F + f];
  const float* nv = nodev + (size_t)b * DA_N * DA_F;
  const float* lv = leafv + (size_t)b * DA_L * DA_F;
  float* up = upward + (size_t)b * DA_L * DA_F;
  float run = 0.f;
  for (int l = DA_L - 1; l >= 0; --l) {
    float x = (r + nv[(size_t)(l >> 3) * DA_F + f] + lv[(size_t)l * DA_F + f]) * (1.0f / 3.0f);
    run += x;
    up[(size_t)l * DA_F + f] = run / (float)(DA_L - l);
  }
}

// ---------------------------------------------------------------------------
// node_hat[b,n,f] = sum_c softmax_c(logits[b,n,:]) * upward[b,n*8+c,f]
// stored TRANSPOSED f16 as [B, F, N] (V-operand for attention).
// ---------------------------------------------------------------------------
__global__ void da_nodehat_kernel(const float* __restrict__ logits,
                                  const float* __restrict__ upward,
                                  _Float16* __restrict__ nodehatT) {
  int idx = blockIdx.x;            // B*N
  int b = idx >> 9;
  int n = idx & (DA_N - 1);
  int f = threadIdx.x;             // 0..63
  const float* lg = logits + (size_t)b * DA_L + n * DA_BR;
  float e[DA_BR];
  float mx = -1e30f;
#pragma unroll
  for (int c = 0; c < DA_BR; ++c) mx = fmaxf(mx, lg[c]);
  float sm = 0.f;
#pragma unroll
  for (int c = 0; c < DA_BR; ++c) { e[c] = __expf(lg[c] - mx); sm += e[c]; }
  const float* up = upward + ((size_t)b * DA_L + n * DA_BR) * DA_F + f;
  float acc = 0.f;
#pragma unroll
  for (int c = 0; c < DA_BR; ++c) acc += e[c] * up[(size_t)c * DA_F];
  nodehatT[((size_t)b * DA_F + f) * DA_N + n] = (_Float16)(acc / sm);
}

// ---------------------------------------------------------------------------
// Cooperative async stage of one 32-key chunk into LDS:
//   K rows:  32 x 64 f16  (4 KB)  -> ksh
//   V^T rows:64 x 32 f16  (4 KB)  -> vsh
// 256 threads x 2 async b128 transfers each.
// ---------------------------------------------------------------------------
__device__ __forceinline__ void da_stage_chunk(const _Float16* __restrict__ kh,
                                               const _Float16* __restrict__ vT,
                                               int RK, int kc,
                                               _Float16* ksh, _Float16* vsh, int tid) {
  {
    int row = tid >> 3, piece = tid & 7;     // 32 rows x 8 pieces of 16 B
    da_async_b128(da_lds_off(ksh + row * DA_F + piece * 8),
                  kh + (size_t)(kc + row) * DA_F + piece * 8);
  }
  {
    int f = tid >> 2, piece = tid & 3;       // 64 rows x 4 pieces of 16 B
    da_async_b128(da_lds_off(vsh + f * 32 + piece * 8),
                  vT + (size_t)f * RK + kc + piece * 8);
  }
}

// ---------------------------------------------------------------------------
// Flash pass over RK keys for one 16-row t-tile (per wave), with the K / V^T
// chunks double-buffered in LDS by the whole 8-wave block.
//   Scores transposed: S^T = K(16xF) * q^T(Fx16) via 2 WMMAs / key-tile.
//   Online softmax over keys (per-lane stats + 1 shfl_xor(16)).
//   out^T accum: acc = V^T(16x32keys) * P^T(32keysx16t) via 4 WMMAs / 32 keys.
// Returns the softmax denominator (un-normalized acc left in acc[]).
// ---------------------------------------------------------------------------
__device__ __forceinline__ float da_flash_pass(const _Float16* __restrict__ kh,  // [RK,F]
                                               const _Float16* __restrict__ vT,  // [F,RK]
                                               int RK, int li, int hf, int tid,
                                               v16h bq0, v16h bq1, v8f acc[4],
                                               _Float16* ksh,   // 2 x 32*64
                                               _Float16* vsh) { // 2 x 64*32
  const float scale = 0.125f;  // 1/sqrt(F)
  float m = -1e30f, s = 0.f;
  const int nch = RK / 32;     // 16 (node) or 128 (leaf): even -> safe handoff

  da_stage_chunk(kh, vT, RK, 0, ksh, vsh, tid);
  for (int ci = 0; ci < nch; ++ci) {
    int p = ci & 1;
    da_wait_async0();          // own async loads for buffer p complete
    __syncthreads();           // whole block's loads done; prior buffer free
    if (ci + 1 < nch) {
      int pn = (ci + 1) & 1;
      da_stage_chunk(kh, vT, RK, (ci + 1) * 32,
                     ksh + pn * (32 * DA_F), vsh + pn * (DA_F * 32), tid);
    }
    const _Float16* K = ksh + p * (32 * DA_F);
    const _Float16* V = vsh + p * (DA_F * 32);

    v8f s0 = {}, s1 = {};
    {
      const _Float16* kr = K + (size_t)li * DA_F + hf * 8;
      v16h a0 = da_cat8(*(const v8h*)kr, *(const v8h*)(kr + 16));
      v16h a1 = da_cat8(*(const v8h*)(kr + 32), *(const v8h*)(kr + 48));
      s0 = da_wmma(a0, bq0, s0);
      s0 = da_wmma(a1, bq1, s0);
    }
    {
      const _Float16* kr = K + (size_t)(16 + li) * DA_F + hf * 8;
      v16h a0 = da_cat8(*(const v8h*)kr, *(const v8h*)(kr + 16));
      v16h a1 = da_cat8(*(const v8h*)(kr + 32), *(const v8h*)(kr + 48));
      s1 = da_wmma(a0, bq0, s1);
      s1 = da_wmma(a1, bq1, s1);
    }

    // online softmax over this 32-key chunk (per t-column = per lane group)
    float p0[8], p1[8];
    float cmax = -1e30f;
#pragma unroll
    for (int j = 0; j < 8; ++j) {
      p0[j] = s0[j] * scale;
      p1[j] = s1[j] * scale;
      cmax = fmaxf(cmax, fmaxf(p0[j], p1[j]));
    }
    cmax = fmaxf(cmax, __shfl_xor(cmax, 16));
    float mnew = fmaxf(m, cmax);
    float corr = __expf(m - mnew);
    float lsum = 0.f;
#pragma unroll
    for (int j = 0; j < 8; ++j) {
      p0[j] = __expf(p0[j] - mnew);
      p1[j] = __expf(p1[j] - mnew);
      lsum += p0[j] + p1[j];
    }
    lsum += __shfl_xor(lsum, 16);
    s = s * corr + lsum;
    m = mnew;
#pragma unroll
    for (int g = 0; g < 4; ++g)
#pragma unroll
      for (int j = 0; j < 8; ++j) acc[g][j] *= corr;

    // Build P^T B-operand (32 keys x 16 t) from the two score C-fragments.
    v16h pb = {};
#pragma unroll
    for (int j = 0; j < 8; ++j) {
      float o0 = __shfl_xor(p0[j], 16);
      float o1 = __shfl_xor(p1[j], 16);
      pb[j]     = (_Float16)(hf ? o1 : p0[j]);  // keys base+0..7
      pb[8 + j] = (_Float16)(hf ? p1[j] : o0);  // keys base+8..15
    }
    // acc += V^T * P^T   (A = V^T rows = features, K = keys)
#pragma unroll
    for (int g = 0; g < 4; ++g) {
      const _Float16* vr = V + (size_t)(g * 16 + li) * 32 + hf * 16;
      v16h av = da_cat8(*(const v8h*)vr, *(const v8h*)(vr + 8));
      acc[g] = da_wmma(av, pb, acc[g]);
    }
  }
  __syncthreads();   // all waves done with the final buffer before reuse
  return s;
}

// ---------------------------------------------------------------------------
// Fused attention: block = 8 waves, all in the same batch; wave = 16 t-rows.
//   out = softmax_F( softmax_N(q nk^T) @ node_hat + softmax_L(q lk^T) @ leaf_v )
// ---------------------------------------------------------------------------
__global__ void __launch_bounds__(256)
da_attn_kernel(const _Float16* __restrict__ qh,       // [B,T,F]
               const _Float16* __restrict__ nodekh,   // [B,N,F]
               const _Float16* __restrict__ leafkh,   // [B,L,F]
               const _Float16* __restrict__ nodehatT, // [B,F,N]
               const _Float16* __restrict__ leafvT,   // [B,F,L]
               float* __restrict__ out) {             // [B,T,F]
  __shared__ __align__(16) _Float16 ksh[2 * 32 * DA_F];  // 8 KB
  __shared__ __align__(16) _Float16 vsh[2 * DA_F * 32];  // 8 KB

  int tid = threadIdx.x;
  int wave = tid >> 5;
  int lane = tid & 31;
  int li = lane & 15;
  int hf = lane >> 4;
  int b = blockIdx.x >> 3;                       // 8 blocks per batch
  int tbase = ((blockIdx.x & 7) * 8 + wave) * 16;

  // q^T B-operand fragments for K=F (two 32-wide chunks)
  const _Float16* qr = qh + ((size_t)b * DA_T + tbase + li) * DA_F + hf * 8;
  v16h bq0 = da_cat8(*(const v8h*)qr, *(const v8h*)(qr + 16));
  v16h bq1 = da_cat8(*(const v8h*)(qr + 32), *(const v8h*)(qr + 48));

  v8f accn[4]; accn[0] = {}; accn[1] = {}; accn[2] = {}; accn[3] = {};
  v8f accl[4]; accl[0] = {}; accl[1] = {}; accl[2] = {}; accl[3] = {};

  float sn = da_flash_pass(nodekh + (size_t)b * DA_N * DA_F,
                           nodehatT + (size_t)b * DA_F * DA_N,
                           DA_N, li, hf, tid, bq0, bq1, accn, ksh, vsh);
  float sl = da_flash_pass(leafkh + (size_t)b * DA_L * DA_F,
                           leafvT + (size_t)b * DA_F * DA_L,
                           DA_L, li, hf, tid, bq0, bq1, accl, ksh, vsh);
  float invn = 1.f / sn;
  float invl = 1.f / sl;

  float tot[4][8];
#pragma unroll
  for (int g = 0; g < 4; ++g)
#pragma unroll
    for (int j = 0; j < 8; ++j) tot[g][j] = accn[g][j] * invn + accl[g][j] * invl;

  // final softmax over F (32 values in own lane + 32 in the other half-wave)
  float fm = -1e30f;
#pragma unroll
  for (int g = 0; g < 4; ++g)
#pragma unroll
    for (int j = 0; j < 8; ++j) fm = fmaxf(fm, tot[g][j]);
  fm = fmaxf(fm, __shfl_xor(fm, 16));
  float fs = 0.f;
#pragma unroll
  for (int g = 0; g < 4; ++g)
#pragma unroll
    for (int j = 0; j < 8; ++j) {
      float e = __expf(tot[g][j] - fm);
      tot[g][j] = e;
      fs += e;
    }
  fs += __shfl_xor(fs, 16);
  float inv = 1.f / fs;

  size_t trow = (size_t)b * DA_T + tbase + li;
#pragma unroll
  for (int g = 0; g < 4; ++g)
#pragma unroll
    for (int j = 0; j < 8; ++j)
      out[trow * DA_F + g * 16 + j + 8 * hf] = tot[g][j] * inv;
}

// ---------------------------------------------------------------------------
// Host launcher
// ---------------------------------------------------------------------------
extern "C" void kernel_launch(void* const* d_in, const int* in_sizes, int n_in,
                              void* d_out, int out_size, void* d_ws, size_t ws_size,
                              hipStream_t stream) {
  const float* root   = (const float*)d_in[0];
  const float* node   = (const float*)d_in[1];
  const float* leaf   = (const float*)d_in[2];
  const float* target = (const float*)d_in[3];
  const float* Wq   = (const float*)d_in[4];
  const float* bq   = (const float*)d_in[5];
  const float* Wk   = (const float*)d_in[6];
  const float* bk   = (const float*)d_in[7];
  const float* Wv   = (const float*)d_in[8];
  const float* bv   = (const float*)d_in[9];
  const float* Wagg = (const float*)d_in[10];
  const float* bagg = (const float*)d_in[11];
  float* out = (float*)d_out;

  char* wsp = (char*)d_ws;
  size_t off = 0;
  auto take = [&](size_t bytes) -> void* {
    void* p = wsp + off;
    off = (off + bytes + 255) & ~(size_t)255;
    return p;
  };
  _Float16* WTq      = (_Float16*)take((size_t)DA_F * DA_D * 2);
  _Float16* WTk      = (_Float16*)take((size_t)DA_F * DA_D * 2);
  _Float16* WTv      = (_Float16*)take((size_t)DA_F * DA_D * 2);
  _Float16* qh       = (_Float16*)take((size_t)DA_B * DA_T * DA_F * 2);
  _Float16* nodekh   = (_Float16*)take((size_t)DA_B * DA_N * DA_F * 2);
  _Float16* leafkh   = (_Float16*)take((size_t)DA_B * DA_L * DA_F * 2);
  float*    nodev    = (float*)   take((size_t)DA_B * DA_N * DA_F * 4);
  float*    leafv    = (float*)   take((size_t)DA_B * DA_L * DA_F * 4);
  _Float16* leafvT   = (_Float16*)take((size_t)DA_B * DA_L * DA_F * 2);
  float*    upward   = (float*)   take((size_t)DA_B * DA_L * DA_F * 4);
  float*    logits   = (float*)   take((size_t)DA_B * DA_L * 4);
  _Float16* nodehatT = (_Float16*)take((size_t)DA_B * DA_N * DA_F * 2);

  // 1) weight transpose+convert (f32 [D,F] -> f16 [F,D])
  da_wcvt_kernel<<<(DA_D * DA_F) / 256, 256, 0, stream>>>(Wq, WTq);
  da_wcvt_kernel<<<(DA_D * DA_F) / 256, 256, 0, stream>>>(Wk, WTk);
  da_wcvt_kernel<<<(DA_D * DA_F) / 256, 256, 0, stream>>>(Wv, WTv);

  // 2) projections (one wave per 16-row tile; 8 waves per block)
  da_proj_kernel<<<(DA_B * DA_T) / 128, 256, 0, stream>>>(
      target, WTq, bq, nullptr, qh, nullptr, DA_B * DA_T);
  da_proj_kernel<<<(DA_B * DA_N) / 128, 256, 0, stream>>>(
      node, WTk, bk, nullptr, nodekh, nullptr, DA_B * DA_N);
  da_proj_kernel<<<(DA_B * DA_L) / 128, 256, 0, stream>>>(
      leaf, WTk, bk, nullptr, leafkh, nullptr, DA_B * DA_L);
  da_proj_kernel<<<(DA_B * DA_N) / 128, 256, 0, stream>>>(
      node, WTv, bv, nodev, nullptr, nullptr, DA_B * DA_N);
  da_proj_kernel<<<(DA_B * DA_L) / 128, 256, 0, stream>>>(
      leaf, WTv, bv, leafv, nullptr, leafvT, DA_B * DA_L);

  // 3) aggregation logits (leaf @ Wagg + bagg)
  da_agg_logits_kernel<<<(DA_B * DA_L) / 8, 256, 0, stream>>>(leaf, Wagg, bagg, logits);

  // 4) tree interp + suffix-mean scan
  da_upward_kernel<<<DA_B, DA_F, 0, stream>>>(root, nodev, leafv, upward);

  // 5) softmax-weighted node aggregation -> node_hat^T (f16)
  da_nodehat_kernel<<<DA_B * DA_N, DA_F, 0, stream>>>(logits, upward, nodehatT);

  // 6) fused dual flash-attention + final softmax over F
  //    64 blocks x 256 threads; 8 blocks per batch, waves stream LDS-staged
  //    K / V^T chunks loaded once per block via async global->LDS DMA.
  da_attn_kernel<<<DA_B * 8, 256, 0, stream>>>(
      qh, nodekh, leafkh, nodehatT, leafvT, out);
}
